// SNN1syn_85925115724561
// MI455X (gfx1250) — matrix operand order repaired and checked by reading, not compiled, not run on hardware
//
#include <hip/hip_runtime.h>
#include <hip/hip_bf16.h>

typedef __attribute__((ext_vector_type(16))) __bf16 v16bf;
typedef __attribute__((ext_vector_type(8)))  __bf16 v8bf;
typedef __attribute__((ext_vector_type(8)))  float  v8f;

#define B_DIM 64
#define T_DIM 256
#define F_DIM 1024
#define N_TILES 64           // F / 16
#define K_TILES 32           // F / 32

// ---------------------------------------------------------------------------
// Prologue 1: zero the t=0 spike read-buffer and the global-barrier counter.
// ---------------------------------------------------------------------------
__global__ __launch_bounds__(256) void snn_init(unsigned short* __restrict__ spk0,
                                                unsigned* __restrict__ ctr) {
    int i = blockIdx.x * 256 + threadIdx.x;
    if (i < B_DIM * F_DIM) spk0[i] = 0;       // bf16 zero
    if (i == 0) *ctr = 0;
}

// ---------------------------------------------------------------------------
// Prologue 2: split W into bf16 hi/lo and pre-swizzle into per-lane B-fragment
// order for v_wmma_f32_16x16x32_bf16.
// B (32x16, K-major): lanes 0-15 hold K=0..15 of column n=lane, lanes 16-31
// hold K=16..31; element e of the 16-elem per-lane vector is K = 16*half + e.
// Destination index: (((kTile*64 + nTile)*32 + lane)*16 + e)
// ---------------------------------------------------------------------------
__global__ __launch_bounds__(256) void snn_pack_w(const float* __restrict__ W,
                                                  __bf16* __restrict__ whi,
                                                  __bf16* __restrict__ wlo) {
    int i     = blockIdx.x * 256 + threadIdx.x;   // 0 .. 1M-1
    int e     = i & 15;
    int lane  = (i >> 4) & 31;
    int frag  = i >> 9;                           // kTile*64 + nTile
    int nTile = frag & (N_TILES - 1);
    int kTile = frag >> 6;
    int half  = lane >> 4;
    int n     = nTile * 16 + (lane & 15);
    int k     = kTile * 32 + half * 16 + e;
    float w   = W[n * F_DIM + k];                 // out[b,n] = sum_k spk[b,k]*W[n,k]
    __bf16 h  = (__bf16)w;
    whi[i] = h;
    wlo[i] = (__bf16)(w - (float)h);
}

// ---------------------------------------------------------------------------
// Persistent SNN kernel: 64 blocks x 256 threads (8 wave32).
//   block  -> N-tile (16 output features)
//   wave   -> (mTile 0..3, kHalf 0..1); kHalf splits the K=1024 reduction
//   state  -> syn/mem live in kHalf=0 waves' registers, per C-tile layout
// Each timestep: 16 K-tiles/wave * 2 WMMA (hi+lo) -> LDS combine -> state
// update -> write spk(bf16) double buffer -> atomic grid barrier.
// ---------------------------------------------------------------------------
__global__ __launch_bounds__(256) void snn_persistent(
    const float* __restrict__ x, const float* __restrict__ bias,
    const __bf16* __restrict__ whi, const __bf16* __restrict__ wlo,
    __bf16* __restrict__ spkbuf,          // [2][B_DIM][F_DIM] bf16
    unsigned* __restrict__ ctr,
    float* __restrict__ spk_out, float* __restrict__ mem_out) {

    const int nTile = blockIdx.x;
    const int wave  = threadIdx.x >> 5;
    const int lane  = threadIdx.x & 31;
    const int mTile = wave & 3;
    const int kHalf = wave >> 2;
    const int half  = lane >> 4;
    const int nIdx  = lane & 15;
    const int fcol  = nTile * 16 + nIdx;

    __shared__ float red[4][8][32];       // kHalf=1 partial C tiles (4 KB)

    const float bval = bias[fcol];

    float syn[8], mem[8];
#pragma unroll
    for (int j = 0; j < 8; ++j) { syn[j] = 0.0f; mem[j] = 0.0f; }

    // A-fragment (16x32 bf16, M-major) addressing for this lane:
    // lane holds row M = lane&15; K = {0..15} (half 0) or {16..31}.
    const int aRow  = mTile * 16 + nIdx;
    const int aOff0 = half ? 16 : 0;
    const int aOff1 = half ? 24 : 8;
    const int kTile0 = kHalf * 16;

    // Per-lane C-tile element (VGPR j): row b = mTile*16 + j + 8*half, col fcol.
    size_t xoff[8];
    int    woff[8];
#pragma unroll
    for (int j = 0; j < 8; ++j) {
        int brow = mTile * 16 + j + 8 * half;
        xoff[j] = (size_t)brow * T_DIM * F_DIM + fcol;   // + t*F_DIM per step
        woff[j] = brow * F_DIM + fcol;
    }

    const size_t bfragBase = (((size_t)kTile0 * N_TILES + nTile) * 32 + lane) * 16;
    const size_t bfragStep = (size_t)N_TILES * 32 * 16;  // stride per kTile

    unsigned target = 0;

    for (int t = 0; t < T_DIM; ++t) {
        const __bf16* spkR = spkbuf + (size_t)(t & 1) * (B_DIM * F_DIM);
        __bf16*       spkW = spkbuf + (size_t)((t + 1) & 1) * (B_DIM * F_DIM);

        v8f acc = {};
        const __bf16* aBase = spkR + (size_t)aRow * F_DIM;
        const __bf16* bh = whi + bfragBase;
        const __bf16* bl = wlo + bfragBase;

#pragma unroll 2
        for (int kt = 0; kt < 16; ++kt) {
            const int k0 = (kTile0 + kt) * 32;
            union { v16bf v; v8bf h[2]; } au;
            au.h[0] = *(const v8bf*)(aBase + k0 + aOff0);   // global_load_b128
            au.h[1] = *(const v8bf*)(aBase + k0 + aOff1);
            v16bf bhv = *(const v16bf*)bh;
            v16bf blv = *(const v16bf*)bl;
            acc = __builtin_amdgcn_wmma_f32_16x16x32_bf16(
                      false, au.v, false, bhv, (short)0, acc, false, false);
            acc = __builtin_amdgcn_wmma_f32_16x16x32_bf16(
                      false, au.v, false, blv, (short)0, acc, false, false);
            bh += bfragStep; bl += bfragStep;
        }

        if (kHalf) {
#pragma unroll
            for (int j = 0; j < 8; ++j) red[mTile][j][lane] = acc[j];
        }
        __syncthreads();
        if (!kHalf) {
#pragma unroll
            for (int j = 0; j < 8; ++j) {
                float rec   = acc[j] + red[mTile][j][lane];
                float reset = (mem[j] > 1.0f) ? 1.0f : 0.0f;     // THR = 1
                size_t oidx = xoff[j] + (size_t)t * F_DIM;
                float s  = 0.8f * syn[j] + x[oidx] + rec + bval; // ALPHA = 0.8
                float m  = 0.9f * mem[j] + s - reset;            // BETA  = 0.9
                float sp = ((m - 1.0f) > 0.0f) ? 1.0f : 0.0f;
                syn[j] = s; mem[j] = m;
                spk_out[oidx] = sp;
                mem_out[oidx] = m;
                spkW[woff[j]] = (__bf16)sp;
            }
        }

        // --- grid-wide barrier: make spkW device-visible, then arrive/wait ---
        __threadfence();                  // release: flush this thread's stores
        __syncthreads();
        target += gridDim.x;
        if (threadIdx.x == 0) {
            __hip_atomic_fetch_add(ctr, 1u, __ATOMIC_RELEASE,
                                   __HIP_MEMORY_SCOPE_AGENT);
            while (__hip_atomic_load(ctr, __ATOMIC_ACQUIRE,
                                     __HIP_MEMORY_SCOPE_AGENT) < target) {
                __builtin_amdgcn_s_sleep(8);
            }
        }
        __syncthreads();
        __threadfence();                  // acquire: invalidate stale lines
    }
}

// ---------------------------------------------------------------------------
// Launch: inputs {x[64,256,1024] f32, W[1024,1024] f32, b[1024] f32}
// Output: spk_rec | mem_rec concatenated, each [64,256,1024] f32.
// Workspace: whi 2MB | wlo 2MB | spk double buffer 256KB | barrier counter.
// ---------------------------------------------------------------------------
extern "C" void kernel_launch(void* const* d_in, const int* in_sizes, int n_in,
                              void* d_out, int out_size, void* d_ws, size_t ws_size,
                              hipStream_t stream) {
    (void)in_sizes; (void)n_in; (void)out_size; (void)ws_size;
    const float* x    = (const float*)d_in[0];
    const float* W    = (const float*)d_in[1];
    const float* bias = (const float*)d_in[2];
    float* spk_out = (float*)d_out;
    float* mem_out = spk_out + (size_t)B_DIM * T_DIM * F_DIM;

    char* ws = (char*)d_ws;
    __bf16*   whi    = (__bf16*)(ws);
    __bf16*   wlo    = (__bf16*)(ws + (2u << 20));
    __bf16*   spkbuf = (__bf16*)(ws + (4u << 20));
    unsigned* ctr    = (unsigned*)(ws + (4u << 20) + (256u << 10));

    snn_init<<<256, 256, 0, stream>>>((unsigned short*)spkbuf, ctr);
    snn_pack_w<<<(F_DIM * F_DIM) / 256, 256, 0, stream>>>(W, whi, wlo);
    snn_persistent<<<N_TILES, 256, 0, stream>>>(x, bias, whi, wlo, spkbuf, ctr,
                                                spk_out, mem_out);
}
